// SetCriterion_83906481094673
// MI455X (gfx1250) — compile-verified, matrix operand order checked by reading, not run on hardware
//
#include <hip/hip_runtime.h>
#include <hip/hip_bf16.h>

typedef __attribute__((ext_vector_type(16))) _Float16 v16h;
typedef __attribute__((ext_vector_type(8)))  float    v8f;

// problem constants (from reference)
#define BB 8
#define NN 5000
#define CC 80
#define MM 64
#define TT 28
#define PIX (TT*TT)
#define NUM_INS (BB*MM)

#define ALPHA_F 0.25f
#define COST_CLS_F 2.0f
#define COST_L1_F 5.0f
#define COST_GIOU_F 2.0f
#define W_CLS_F 2.0f
#define W_L1_F 5.0f
#define W_GIOU_F 2.0f
#define W_MASK_F 5.0f
#define EPS_F 1e-8f

#define CLS_BLOCKS 512

__device__ __forceinline__ float sigmoidf_(float x) { return 1.0f / (1.0f + __expf(-x)); }

// pos_cost - neg_cost for one logit (focal matching cost before class gather)
__device__ __forceinline__ float focal_diff(float x) {
    float p   = sigmoidf_(x);
    float omp = 1.0f - p;
    float pos = -__logf(p + EPS_F)   * ALPHA_F          * omp * omp;
    float neg = -__logf(omp + EPS_F) * (1.0f - ALPHA_F) * p * p;
    return pos - neg;
}

// monotonic float -> uint key (works for negatives; argmin order preserved)
__device__ __forceinline__ unsigned int fkey(float f) {
    unsigned int u = __float_as_uint(f);
    return (u & 0x80000000u) ? ~u : (u | 0x80000000u);
}

__device__ __forceinline__ unsigned long long ullmin_(unsigned long long a, unsigned long long b) {
    return a < b ? a : b;
}

// ---------------------------------------------------------------- init
__global__ void initKernel(unsigned long long* __restrict__ amin, int* __restrict__ mcls) {
    int i = blockIdx.x * blockDim.x + threadIdx.x;
    if (i < BB * MM) amin[i] = 0xFFFFFFFFFFFFFFFFull;
    if (i < BB * NN) mcls[i] = -1;
}

// ---------------------------------------------------------------- matching
// One wave (32 lanes) handles a 16-row n-tile for batch b.
// Class cost = diff(16x80,f16) x OneHot(80x64,f16) via 12x v_wmma_f32_16x16x32_f16
// (K padded 80->96). VALU computes L1/GIoU/center for the same tile; the XDL
// WMMA co-executes with it (ISA: 16-bit WMMA tracked as TRANS).
__global__ __launch_bounds__(256) void matchKernel(
    const float*  __restrict__ logits,   // [B,N,C]
    const float4* __restrict__ pboxes,   // [B,N]
    const int*    __restrict__ gtc,      // [B,M]
    const float4* __restrict__ gboxes,   // [B,M]
    const float4* __restrict__ isz,      // [B]
    unsigned long long* __restrict__ amin)  // [B,M] packed (costkey<<32)|n
{
    const int lane  = threadIdx.x & 31;
    const int wave  = threadIdx.x >> 5;
    const int b     = blockIdx.y;
    const int NT    = (NN + 15) / 16;           // 313 tiles
    const int tile  = blockIdx.x * 8 + wave;
    if (tile >= NT) return;                     // wave-uniform exit; EXEC stays all-1

    const int n0    = tile * 16;
    const int rlane = lane & 15;
    const int hi    = (lane >= 16) ? 8 : 0;
    const int naddr = min(n0 + rlane, NN - 1);
    const float* rowp = logits + (size_t)(b * NN + naddr) * CC;

    // A fragments: lane holds row (n0+rlane); half h -> K per 16-bit A layout
    v16h A[3];
#pragma unroll
    for (int kc = 0; kc < 3; ++kc) {
        const int kbase = kc * 32;
#pragma unroll
        for (int h = 0; h < 16; ++h) {
            int k = ((h < 8) ? h : (h + 8)) + hi + kbase;
            float d = (k < CC) ? focal_diff(rowp[k]) : 0.0f;
            A[kc][h] = (_Float16)d;
        }
    }

    // gt classes for this lane's output columns (m = mc*16 + rlane)
    int clsm[4];
#pragma unroll
    for (int mc = 0; mc < 4; ++mc) clsm[mc] = gtc[b * MM + mc * 16 + rlane];

    // 4 accumulators: [16 n-rows x 16 m-cols] each
    v8f acc[4];
#pragma unroll
    for (int mc = 0; mc < 4; ++mc) {
        v8f c = {};
#pragma unroll
        for (int kc = 0; kc < 3; ++kc) {
            const int kbase = kc * 32;
            v16h Bf;
#pragma unroll
            for (int h = 0; h < 16; ++h) {
                int k = ((h < 8) ? h : (h + 8)) + hi + kbase;
                Bf[h] = (_Float16)((k == clsm[mc]) ? 1.0f : 0.0f);
            }
            c = __builtin_amdgcn_wmma_f32_16x16x32_f16(
                    false, A[kc], false, Bf, (short)0, c, false, false);
        }
        acc[mc] = c;
    }

    // Scalar part: boxes for this lane's 8 rows (n = n0 + r + hi)
    const float4 im = isz[b];
    float4 nb[8];
    float  cx[8], cy[8];
#pragma unroll
    for (int r = 0; r < 8; ++r) {
        int na = min(n0 + r + hi, NN - 1);
        float4 p4 = pboxes[b * NN + na];
        nb[r] = make_float4(p4.x / im.x, p4.y / im.y, p4.z / im.z, p4.w / im.w);
        cx[r] = 0.5f * (p4.x + p4.z);
        cy[r] = 0.5f * (p4.y + p4.w);
    }

#pragma unroll
    for (int mc = 0; mc < 4; ++mc) {
        const int m = mc * 16 + rlane;
        const float4 g4 = gboxes[b * MM + m];
        const float4 ng = make_float4(g4.x / im.x, g4.y / im.y, g4.z / im.z, g4.w / im.w);
        const float area_b = (ng.z - ng.x) * (ng.w - ng.y);
        unsigned long long best = 0xFFFFFFFFFFFFFFFFull;
#pragma unroll
        for (int r = 0; r < 8; ++r) {
            const int n = n0 + r + hi;
            const float4 a = nb[r];
            float l1 = fabsf(a.x - ng.x) + fabsf(a.y - ng.y) +
                       fabsf(a.z - ng.z) + fabsf(a.w - ng.w);
            float area_a = (a.z - a.x) * (a.w - a.y);
            float iw = fmaxf(fminf(a.z, ng.z) - fmaxf(a.x, ng.x), 0.0f);
            float ih = fmaxf(fminf(a.w, ng.w) - fmaxf(a.y, ng.y), 0.0f);
            float inter = iw * ih;
            float uni = area_a + area_b - inter;
            float iou = inter / (uni + EPS_F);
            float ew = fmaxf(fmaxf(a.z, ng.z) - fminf(a.x, ng.x), 0.0f);
            float eh = fmaxf(fmaxf(a.w, ng.w) - fminf(a.y, ng.y), 0.0f);
            float ae = ew * eh;
            float giou = iou - (ae - uni) / (ae + EPS_F);
            bool inc = (cx[r] > g4.x) && (cx[r] < g4.z) && (cy[r] > g4.y) && (cy[r] < g4.w);
            float cost = COST_CLS_F * acc[mc][r] + COST_L1_F * l1 +
                         COST_GIOU_F * (1.0f - giou) + (inc ? 0.0f : 100000.0f);
            if (n < NN) {
                unsigned long long key =
                    ((unsigned long long)fkey(cost) << 32) | (unsigned int)n;
                best = ullmin_(best, key);
            }
        }
        // lanes L and L^16 share the same m; combine, then one atomic per pair
        unsigned int lo = (unsigned int)best, hw = (unsigned int)(best >> 32);
        lo = __shfl_xor(lo, 16, 32);
        hw = __shfl_xor(hw, 16, 32);
        best = ullmin_(best, ((unsigned long long)hw << 32) | lo);
        if (lane < 16) atomicMin(&amin[b * MM + m], best);
    }
}

// ---------------------------------------------------------------- scatter (last-write-wins over m, like .at[].set)
__global__ void scatterKernel(const unsigned long long* __restrict__ amin,
                              const int* __restrict__ gtc, int* __restrict__ mcls) {
    int b = threadIdx.x;
    if (b >= BB) return;
    for (int m = 0; m < MM; ++m) {
        int n = (int)(amin[b * MM + m] & 0xFFFFFFFFu);
        mcls[b * NN + n] = gtc[b * MM + m];
    }
}

__device__ __forceinline__ float blockReduce256(float v, float* sm) {
    sm[threadIdx.x] = v;
    __syncthreads();
    for (int s = 128; s > 0; s >>= 1) {
        if ((int)threadIdx.x < s) sm[threadIdx.x] += sm[threadIdx.x + s];
        __syncthreads();
    }
    float r = sm[0];
    __syncthreads();
    return r;
}

// ---------------------------------------------------------------- focal classification loss (partials per block)
__global__ __launch_bounds__(256) void clsLossKernel(const float* __restrict__ logits,
                                                     const int* __restrict__ mcls,
                                                     float* __restrict__ partials) {
    __shared__ float sm[256];
    const long long total = (long long)BB * NN * CC;
    const long long stride = (long long)gridDim.x * 256;
    float sum = 0.0f;
    for (long long i = (long long)blockIdx.x * 256 + threadIdx.x; i < total; i += stride) {
        __builtin_prefetch(&logits[i + stride], 0, 0);   // global_prefetch_b8
        int c = (int)(i % CC);
        long long bn = i / CC;
        float t = (mcls[bn] == c) ? 1.0f : 0.0f;
        float x = logits[i];
        float p = sigmoidf_(x);
        float ce = fmaxf(x, 0.0f) - x * t + __logf(1.0f + __expf(-fabsf(x)));
        float pt = p * t + (1.0f - p) * (1.0f - t);
        float at = ALPHA_F * t + (1.0f - ALPHA_F) * (1.0f - t);
        float om = 1.0f - pt;
        sum += at * ce * om * om;
    }
    float r = blockReduce256(sum, sm);
    if (threadIdx.x == 0) partials[blockIdx.x] = r;
}

// ---------------------------------------------------------------- matched-box L1 + GIoU terms
__global__ void boxKernel(const float4* __restrict__ pboxes, const float4* __restrict__ gboxes,
                          const float4* __restrict__ isz, const unsigned long long* __restrict__ amin,
                          float* __restrict__ l1t, float* __restrict__ giout) {
    int i = blockIdx.x * blockDim.x + threadIdx.x;
    if (i >= NUM_INS) return;
    int b = i / MM;
    int n = (int)(amin[i] & 0xFFFFFFFFu);
    float4 im = isz[b];
    float4 p4 = pboxes[b * NN + n];
    float4 g4 = gboxes[i];
    float4 a = make_float4(p4.x / im.x, p4.y / im.y, p4.z / im.z, p4.w / im.w);
    float4 g = make_float4(g4.x / im.x, g4.y / im.y, g4.z / im.z, g4.w / im.w);
    l1t[i] = fabsf(a.x - g.x) + fabsf(a.y - g.y) + fabsf(a.z - g.z) + fabsf(a.w - g.w);
    float area_a = (a.z - a.x) * (a.w - a.y);
    float area_b = (g.z - g.x) * (g.w - g.y);
    float iw = fmaxf(fminf(a.z, g.z) - fmaxf(a.x, g.x), 0.0f);
    float ih = fmaxf(fminf(a.w, g.w) - fmaxf(a.y, g.y), 0.0f);
    float inter = iw * ih;
    float uni = area_a + area_b - inter;
    float iou = inter / (uni + EPS_F);
    float ew = fmaxf(fmaxf(a.z, g.z) - fminf(a.x, g.x), 0.0f);
    float eh = fmaxf(fmaxf(a.w, g.w) - fminf(a.y, g.y), 0.0f);
    float ae = ew * eh;
    giout[i] = 1.0f - (iou - (ae - uni) / (ae + EPS_F));
}

// ---------------------------------------------------------------- dice mask loss (one block per matched instance)
__global__ __launch_bounds__(256) void maskKernel(const float* __restrict__ mlogits,
                                                  const float* __restrict__ gmasks,
                                                  const unsigned long long* __restrict__ amin,
                                                  float* __restrict__ dice) {
    __shared__ float sm[256];
    int inst = blockIdx.x;
    int b = inst / MM;
    int n = (int)(amin[inst] & 0xFFFFFFFFu);
    const float* mp = mlogits + (size_t)(b * NN + n) * PIX;
    const float* gp = gmasks + (size_t)inst * PIX;
    float ai = 0.0f, am = 0.0f, ag = 0.0f;
    for (int p = threadIdx.x; p < PIX; p += 256) {
        float pm = sigmoidf_(mp[p]);
        float g = gp[p];
        ai += pm * g;
        am += pm;
        ag += g;
    }
    float ri = blockReduce256(ai, sm);
    float rm = blockReduce256(am, sm);
    float rg = blockReduce256(ag, sm);
    if (threadIdx.x == 0) dice[inst] = 1.0f - 2.0f * ri / (rm + rg + EPS_F);
}

// ---------------------------------------------------------------- final deterministic combine
__global__ __launch_bounds__(256) void finalKernel(const float* __restrict__ clsp,
                                                   const float* __restrict__ l1t,
                                                   const float* __restrict__ giout,
                                                   const float* __restrict__ dice,
                                                   float* __restrict__ out) {
    __shared__ float sm[256];
    float s;

    s = 0.0f;
    for (int i = threadIdx.x; i < CLS_BLOCKS; i += 256) s += clsp[i];
    float cls_sum = blockReduce256(s, sm);

    s = 0.0f;
    for (int i = threadIdx.x; i < NUM_INS; i += 256) s += l1t[i];
    float l1_sum = blockReduce256(s, sm);

    s = 0.0f;
    for (int i = threadIdx.x; i < NUM_INS; i += 256) s += giout[i];
    float giou_sum = blockReduce256(s, sm);

    s = 0.0f;
    for (int i = threadIdx.x; i < NUM_INS; i += 256) s += dice[i];
    float dice_sum = blockReduce256(s, sm);

    if (threadIdx.x == 0) {
        const float ninv = 1.0f / (float)NUM_INS;
        out[0] = W_CLS_F * cls_sum * ninv;
        out[1] = W_L1_F * l1_sum * ninv;
        out[2] = W_GIOU_F * giou_sum * ninv;
        out[3] = W_MASK_F * dice_sum * ninv;
    }
}

extern "C" void kernel_launch(void* const* d_in, const int* in_sizes, int n_in,
                              void* d_out, int out_size, void* d_ws, size_t ws_size,
                              hipStream_t stream) {
    const float*  logits  = (const float*)d_in[0];   // pred_logits [B,N,C]
    const float4* pboxes  = (const float4*)d_in[1];  // pred_boxes [B,N,4]
    const float*  mlogits = (const float*)d_in[2];   // pred_mask_logits [B,N,T,T]
    const int*    gtc     = (const int*)d_in[3];     // gt_classes [B,M]
    const float4* gboxes  = (const float4*)d_in[4];  // gt_boxes [B,M,4]
    const float*  gmasks  = (const float*)d_in[5];   // gt_masks [B,M,T,T]
    const float4* isz     = (const float4*)d_in[6];  // image_size [B,4]
    float* out = (float*)d_out;

    char* ws = (char*)d_ws;
    unsigned long long* amin = (unsigned long long*)ws;           // 512 * 8   = 4096 B
    int*   mcls  = (int*)(ws + 4096);                             // 40000 * 4 = 160000 B
    float* clsp  = (float*)(ws + 4096 + 160000);                  // 512 floats
    float* l1t   = clsp + CLS_BLOCKS;                             // 512 floats
    float* giout = l1t + NUM_INS;                                 // 512 floats
    float* dice  = giout + NUM_INS;                               // 512 floats

    initKernel<<<(BB * NN + 255) / 256, 256, 0, stream>>>(amin, mcls);

    dim3 mg((((NN + 15) / 16) + 7) / 8, BB);                      // 40 x 8 blocks, 8 waves each
    matchKernel<<<mg, 256, 0, stream>>>(logits, pboxes, gtc, gboxes, isz, amin);

    scatterKernel<<<1, BB, 0, stream>>>(amin, gtc, mcls);
    clsLossKernel<<<CLS_BLOCKS, 256, 0, stream>>>(logits, mcls, clsp);
    boxKernel<<<(NUM_INS + 255) / 256, 256, 0, stream>>>(pboxes, gboxes, isz, amin, l1t, giout);
    maskKernel<<<NUM_INS, 256, 0, stream>>>(mlogits, gmasks, amin, dice);
    finalKernel<<<1, 256, 0, stream>>>(clsp, l1t, giout, dice, out);
}
